// GATblock_43757126811984
// MI455X (gfx1250) — compile-verified, matrix operand order checked by reading, not run on hardware
//
#include <hip/hip_runtime.h>
#include <math.h>

// ---------------------------------------------------------------------------
// GAT (3-layer) for MI455X / gfx1250.
//   - Dense node transforms: v_wmma_f32_16x16x32_f16, wave32, one wave per
//     16-row slab computing all M/16 column tiles (A-fragment reuse).
//     M and K are template constants -> fully unrolled K loop, immediate
//     store offsets, branch-free epilogue in the common (full-tile) case.
//   - W pre-swizzled into WMMA B-fragment order (contiguous b128 loads).
//   - A fragments: contiguous float4 loads + cvt_pk_f16_f32 (no exec masks).
//   - Edge-feature attention term collapsed to eattr @ (We @ a_e^T)  (8xH).
//   - Edge softmax via atomicMax (order-preserving uint encode) + atomicAdd.
//   - Aggregation: float4 gather + 4x global_atomic_add_f32 per chunk.
//   - Conv biases skipped (cancel exactly under BatchNorm).
// Input order (setup_inputs dict order, params recursed in insertion order):
//   0:x 1:edge_weigth
//   2..7  : conv1 {W,We,a_src,a_dst,a_e,b}
//   8..13 : conv2 {...}  14..19 : conv3 {...}
//   20,21 : bn1{g,b} 22,23 : bn2 24,25 : bn3
//   26    : edge_index (int32, [2,E])
// ---------------------------------------------------------------------------

typedef __attribute__((ext_vector_type(16))) _Float16 v16h;
typedef __attribute__((ext_vector_type(8)))  float    v8f;

#define LRELU_SLOPE 0.2f
#define BN_EPS 1e-5f
#define CDIM 16            // per-head channel count (HID == OUT == 16)

__device__ __forceinline__ unsigned enc_f32(float f) {
  unsigned u = __float_as_uint(f);
  return (u & 0x80000000u) ? ~u : (u | 0x80000000u);
}
__device__ __forceinline__ float dec_f32(unsigned u) {
  return (u & 0x80000000u) ? __uint_as_float(u & 0x7FFFFFFFu)
                           : __uint_as_float(~u);
}

// ---------------------------------------------------------------------------
// Pack W[K x M] (f32, row-major) into WMMA B-fragment order, f16:
//   Wp[(tile_n*ksteps + kt)*32 + lane] = v16h fragment for that lane.
// Per ISA 7.12.2 (16-bit 16x32 layout): element e of lane ->
//   kk = kt*32 + e + 8*(lane>>4) + (e>=8 ? 8 : 0), col = tile_n*16 + (lane&15)
// ---------------------------------------------------------------------------
__global__ void pack_w_kernel(const float* __restrict__ W, v16h* __restrict__ Wp,
                              int K, int M) {
  const int t = blockIdx.x * blockDim.x + threadIdx.x;
  const int ksteps = K >> 5;
  const int total = (M >> 4) * ksteps * 32;
  if (t >= total) return;
  const int lane   = t & 31;
  const int kt     = (t >> 5) % ksteps;
  const int tile_n = (t >> 5) / ksteps;
  const int half   = lane >> 4;
  const int col    = tile_n * 16 + (lane & 15);
  v16h b;
#pragma unroll
  for (int e = 0; e < 16; ++e) {
    const int kk = kt * 32 + e + half * 8 + ((e >= 8) ? 8 : 0);
    b[e] = (_Float16)W[(size_t)kk * M + col];
  }
  Wp[t] = b;
}

// ---------------------------------------------------------------------------
// WMMA GEMM: Hout[N x M] = X[N x K] @ W[K x M]; M = NT*16, K = KSTEPS*32.
// One wave per 16-row slab; A loaded as 4x float4 (two contiguous 8-float
// runs per fragment), B fragments prepacked & contiguous.
// ---------------------------------------------------------------------------
template <int NT, int KSTEPS>
__global__ __launch_bounds__(32)
void gemm_wmma_f16(const float* __restrict__ X, const v16h* __restrict__ Wp,
                   float* __restrict__ Hout, int nrows) {
  constexpr int M = NT * 16;
  constexpr int K = KSTEPS * 32;
  const int tile_m = blockIdx.x;
  const int lane   = threadIdx.x;
  const int half   = lane >> 4;
  const int r      = lane & 15;
  int row = tile_m * 16 + r;
  if (row > nrows - 1) row = nrows - 1;       // clamp: no exec divergence
  const float* Xrow = X + (size_t)row * K;

  v8f acc[NT];
#pragma unroll
  for (int t = 0; t < NT; ++t) acc[t] = (v8f){};

#pragma unroll
  for (int kt = 0; kt < KSTEPS; ++kt) {
    const int k0 = kt * 32;
    if (kt + 1 < KSTEPS)
      __builtin_prefetch(Xrow + k0 + 32, 0, 1);           // global_prefetch_b8
    const float4* p0 = (const float4*)(Xrow + k0 + 8 * half);        // e=0..7
    const float4* p1 = (const float4*)(Xrow + k0 + 16 + 8 * half);   // e=8..15
    const float4 x0 = p0[0], x1 = p0[1];
    const float4 x2 = p1[0], x3 = p1[1];
    const float av[16] = {x0.x, x0.y, x0.z, x0.w, x1.x, x1.y, x1.z, x1.w,
                          x2.x, x2.y, x2.z, x2.w, x3.x, x3.y, x3.z, x3.w};
    v16h a;
#pragma unroll
    for (int e = 0; e < 16; ++e) a[e] = (_Float16)av[e];
#pragma unroll
    for (int t = 0; t < NT; ++t) {
      const v16h b = Wp[(size_t)(t * KSTEPS + kt) * 32 + lane];
      acc[t] = __builtin_amdgcn_wmma_f32_16x16x32_f16(false, a, false, b,
                                                      (short)0, acc[t],
                                                      false, false);
    }
  }

  // Epilogue: C/D layout -> lane (half,r) row = i + 8*half, col = r.
  if (tile_m * 16 + 16 <= nrows) {            // full tile: branch-free stores
#pragma unroll
    for (int t = 0; t < NT; ++t) {
      float* p = Hout + (size_t)(tile_m * 16 + half * 8) * M + t * 16 + r;
#pragma unroll
      for (int i = 0; i < 8; ++i) p[i * M] = acc[t][i];   // immediate offsets
    }
  } else {
#pragma unroll
    for (int t = 0; t < NT; ++t) {
#pragma unroll
      for (int i = 0; i < 8; ++i) {
        const int orow = tile_m * 16 + i + half * 8;
        if (orow < nrows)
          Hout[(size_t)orow * M + t * 16 + r] = acc[t][i];
      }
    }
  }
}

// --- self-loop attr: sum incoming eattr + degree ---------------------------
__global__ void loop_sum_kernel(const int* __restrict__ dst,
                                const float* __restrict__ ew,
                                float* __restrict__ lsum,
                                float* __restrict__ cnt, int E) {
  int i = blockIdx.x * blockDim.x + threadIdx.x;
  if (i >= E) return;
  const int d = dst[i];
  atomicAdd(&cnt[d], 1.0f);
  const float* e = ew + (size_t)i * 8;
#pragma unroll
  for (int j = 0; j < 8; ++j) atomicAdd(&lsum[(size_t)d * 8 + j], e[j]);
}

__global__ void loop_div_kernel(float* __restrict__ lsum,
                                const float* __restrict__ cnt, int N) {
  int i = blockIdx.x * blockDim.x + threadIdx.x;
  if (i >= N * 8) return;
  lsum[i] /= fmaxf(cnt[i >> 3], 1.0f);
}

// --- wv[d,h] = sum_c We[d, h*C+c] * a_e[h,c]  (8 x H, tiny) ----------------
__global__ void wv_kernel(const float* __restrict__ We,
                          const float* __restrict__ ae,
                          float* __restrict__ wv, int Hh, int M) {
  int t = threadIdx.x;
  if (t >= 8 * Hh) return;
  const int d = t / Hh, h = t % Hh;
  float s = 0.f;
#pragma unroll
  for (int c = 0; c < CDIM; ++c) s += We[d * M + h * CDIM + c] * ae[h * CDIM + c];
  wv[d * Hh + h] = s;
}

// --- per-node attention terms a_s, a_d -------------------------------------
__global__ void att_node_kernel(const float* __restrict__ hfeat,
                                const float* __restrict__ a_src,
                                const float* __restrict__ a_dst,
                                float* __restrict__ as_, float* __restrict__ ad_,
                                int N, int Hh) {
  int t = blockIdx.x * blockDim.x + threadIdx.x;
  if (t >= N * Hh) return;
  const int n = t / Hh, h = t % Hh;
  const float* hp = hfeat + (size_t)n * Hh * CDIM + h * CDIM;
  float s = 0.f, s2 = 0.f;
#pragma unroll
  for (int c = 0; c < CDIM; ++c) {
    float v = hp[c];
    s  += v * a_src[h * CDIM + c];
    s2 += v * a_dst[h * CDIM + c];
  }
  as_[t] = s;
  ad_[t] = s2;
}

// --- edge pass 1: leaky-relu logits + segment max over dst -----------------
__global__ void edge_logit_max(const int* __restrict__ src, const int* __restrict__ dstv,
                               const float* __restrict__ ew,
                               const float* __restrict__ loop_attr,
                               const float* __restrict__ wv,
                               const float* __restrict__ as_, const float* __restrict__ ad_,
                               float* __restrict__ alpha, unsigned* __restrict__ amax,
                               int E, int N, int Hh) {
  int i = blockIdx.x * blockDim.x + threadIdx.x;
  if (i >= E + N) return;
  int s, d;
  const float* ep;
  if (i < E) { s = src[i]; d = dstv[i]; ep = ew + (size_t)i * 8; }
  else       { s = d = i - E;           ep = loop_attr + (size_t)(i - E) * 8; }
  float ev[8];
#pragma unroll
  for (int j = 0; j < 8; ++j) ev[j] = ep[j];
  for (int h = 0; h < Hh; ++h) {
    float ae = 0.f;
#pragma unroll
    for (int j = 0; j < 8; ++j) ae += ev[j] * wv[j * Hh + h];
    float v = as_[(size_t)s * Hh + h] + ad_[(size_t)d * Hh + h] + ae;
    v = (v > 0.f) ? v : LRELU_SLOPE * v;
    alpha[(size_t)i * Hh + h] = v;
    atomicMax(&amax[(size_t)d * Hh + h], enc_f32(v));
  }
}

// --- edge pass 2: exp(logit - max) + segment sum ---------------------------
__global__ void edge_exp_sum(const int* __restrict__ dstv,
                             float* __restrict__ alpha,
                             const unsigned* __restrict__ amax,
                             float* __restrict__ denom, int E, int N, int Hh) {
  int i = blockIdx.x * blockDim.x + threadIdx.x;
  if (i >= E + N) return;
  const int d = (i < E) ? dstv[i] : (i - E);
  for (int h = 0; h < Hh; ++h) {
    float v = __expf(alpha[(size_t)i * Hh + h] - dec_f32(amax[(size_t)d * Hh + h]));
    alpha[(size_t)i * Hh + h] = v;
    atomicAdd(&denom[(size_t)d * Hh + h], v);
  }
}

// --- edge pass 3: weighted gather/scatter aggregation ----------------------
// One thread per (edge, 4-float chunk); chunk_shift = log2(M/4).
__global__ void edge_aggregate(const int* __restrict__ src, const int* __restrict__ dstv,
                               const float* __restrict__ hfeat,
                               const float* __restrict__ alpha,
                               const float* __restrict__ denom,
                               float* __restrict__ agg,
                               int E, int N, int Hh, int M, int chunk_shift) {
  long long t = (long long)blockIdx.x * blockDim.x + threadIdx.x;
  const int chunks = 1 << chunk_shift;
  const long long tot = (long long)(E + N) * chunks;
  if (t >= tot) return;
  const int i = (int)(t >> chunk_shift);
  const int q = (int)(t & (chunks - 1));
  int s, d;
  if (i < E) { s = src[i]; d = dstv[i]; } else { s = d = i - E; }
  const int head = (q * 4) / CDIM;
  const float w = alpha[(size_t)i * Hh + head] /
                  (denom[(size_t)d * Hh + head] + 1e-16f);
  const float4 v = *(const float4*)(hfeat + (size_t)s * M + q * 4);
  float* out = agg + (size_t)d * M + q * 4;
  atomicAdd(out + 0, v.x * w);
  atomicAdd(out + 1, v.y * w);
  atomicAdd(out + 2, v.z * w);
  atomicAdd(out + 3, v.w * w);
}

// --- BatchNorm statistics (block = M threads, each owns one feature) -------
__global__ void bn_stats(const float* __restrict__ x, float* __restrict__ bsum,
                         float* __restrict__ bsq, int N, int M, int rows_per_block) {
  const int f  = threadIdx.x;
  const int n0 = blockIdx.x * rows_per_block;
  int n1 = n0 + rows_per_block;
  if (n1 > N) n1 = N;
  float s = 0.f, q = 0.f;
  for (int n = n0; n < n1; ++n) {
    float v = x[(size_t)n * M + f];
    s += v; q += v * v;
  }
  atomicAdd(&bsum[f], s);
  atomicAdd(&bsq[f], q);
}

__global__ void bn_apply_relu(const float* __restrict__ x,
                              const float* __restrict__ bsum,
                              const float* __restrict__ bsq,
                              const float* __restrict__ g,
                              const float* __restrict__ bb,
                              float* __restrict__ out, int N, int M) {
  long long t = (long long)blockIdx.x * blockDim.x + threadIdx.x;
  if (t >= (long long)N * M) return;
  const int f = (int)(t % M);
  const float invN = 1.0f / (float)N;
  const float mean = bsum[f] * invN;
  const float var  = bsq[f] * invN - mean * mean;
  const float sc   = g[f] * rsqrtf(var + BN_EPS);
  out[t] = fmaxf(x[t] * sc + (bb[f] - mean * sc), 0.0f);
}

// ---------------------------------------------------------------------------
extern "C" void kernel_launch(void* const* d_in, const int* in_sizes, int n_in,
                              void* d_out, int out_size, void* d_ws, size_t ws_size,
                              hipStream_t stream) {
  const float* x  = (const float*)d_in[0];
  const float* ew = (const float*)d_in[1];
  const float* cW[3]  = {(const float*)d_in[2],  (const float*)d_in[8],  (const float*)d_in[14]};
  const float* cWe[3] = {(const float*)d_in[3],  (const float*)d_in[9],  (const float*)d_in[15]};
  const float* cAs[3] = {(const float*)d_in[4],  (const float*)d_in[10], (const float*)d_in[16]};
  const float* cAd[3] = {(const float*)d_in[5],  (const float*)d_in[11], (const float*)d_in[17]};
  const float* cAe[3] = {(const float*)d_in[6],  (const float*)d_in[12], (const float*)d_in[18]};
  const float* bnG[3] = {(const float*)d_in[20], (const float*)d_in[22], (const float*)d_in[24]};
  const float* bnB[3] = {(const float*)d_in[21], (const float*)d_in[23], (const float*)d_in[25]};
  const int* eidx = (const int*)d_in[26];

  const int Nn = in_sizes[0] / 128;   // 50000
  const int Ee = in_sizes[1] / 8;     // 1.6M
  const int* srcv = eidx;
  const int* dstv = eidx + Ee;

  // --- workspace layout (bump allocator, 256B aligned) ---
  char* base = (char*)d_ws;
  size_t off = 0;
  auto alloc = [&](size_t bytes) -> void* {
    void* p = base + off;
    off = (off + bytes + 255) & ~(size_t)255;
    return p;
  };
  float*    loop_attr = (float*)alloc((size_t)Nn * 8 * 4);
  float*    cnt       = (float*)alloc((size_t)Nn * 4);
  float*    wv        = (float*)alloc(8 * 4 * 4);
  v16h*     wpack     = (v16h*)alloc(4 * 4 * 32 * 32);   // max (M/16)*(K/32)*32 frags
  float*    a_s       = (float*)alloc((size_t)Nn * 4 * 4);
  float*    a_d       = (float*)alloc((size_t)Nn * 4 * 4);
  unsigned* amax      = (unsigned*)alloc((size_t)Nn * 4 * 4);
  float*    denom     = (float*)alloc((size_t)Nn * 4 * 4);
  float*    alpha     = (float*)alloc((size_t)(Ee + Nn) * 4 * 4);
  float*    bufH      = (float*)alloc((size_t)Nn * 64 * 4);   // transformed feats
  float*    bufA      = (float*)alloc((size_t)Nn * 64 * 4);   // aggregation
  float*    bufB      = (float*)alloc((size_t)Nn * 64 * 4);   // activated feats
  float*    bnsum     = (float*)alloc(64 * 4);
  float*    bnsq      = (float*)alloc(64 * 4);
  (void)ws_size; (void)n_in; (void)out_size;

  // --- self-loop attrs (fill_value='mean') ---
  hipMemsetAsync(loop_attr, 0, (size_t)Nn * 8 * 4, stream);
  hipMemsetAsync(cnt,       0, (size_t)Nn * 4,     stream);
  loop_sum_kernel<<<(Ee + 255) / 256, 256, 0, stream>>>(dstv, ew, loop_attr, cnt, Ee);
  loop_div_kernel<<<(Nn * 8 + 255) / 256, 256, 0, stream>>>(loop_attr, cnt, Nn);

  auto run_layer = [&](const float* xin, int K, int Hh, int M, int li, float* outp) {
    // 1) node transform via WMMA (pack W, then one wave per 16-row slab)
    const int ksteps = K >> 5;
    const int packn  = (M >> 4) * ksteps * 32;
    pack_w_kernel<<<(packn + 255) / 256, 256, 0, stream>>>(cW[li], wpack, K, M);
    const int gx = (Nn + 15) / 16;
    if (M == 64 && K == 128)
      gemm_wmma_f16<4, 4><<<gx, 32, 0, stream>>>(xin, wpack, bufH, Nn);
    else if (M == 64)
      gemm_wmma_f16<4, 2><<<gx, 32, 0, stream>>>(xin, wpack, bufH, Nn);
    else
      gemm_wmma_f16<1, 2><<<gx, 32, 0, stream>>>(xin, wpack, bufH, Nn);
    // 2) collapse edge-attention weights
    wv_kernel<<<1, 64, 0, stream>>>(cWe[li], cAe[li], wv, Hh, M);
    // 3) per-node attention terms
    const int nh = Nn * Hh;
    att_node_kernel<<<(nh + 255) / 256, 256, 0, stream>>>(bufH, cAs[li], cAd[li],
                                                          a_s, a_d, Nn, Hh);
    // 4) edge softmax
    hipMemsetAsync(amax,  0, (size_t)nh * 4, stream);
    hipMemsetAsync(denom, 0, (size_t)nh * 4, stream);
    const int tot = Ee + Nn;
    edge_logit_max<<<(tot + 255) / 256, 256, 0, stream>>>(srcv, dstv, ew, loop_attr, wv,
                                                          a_s, a_d, alpha, amax, Ee, Nn, Hh);
    edge_exp_sum<<<(tot + 255) / 256, 256, 0, stream>>>(dstv, alpha, amax, denom, Ee, Nn, Hh);
    // 5) aggregation
    hipMemsetAsync(bufA, 0, (size_t)Nn * M * 4, stream);
    const int cshift = (M == 64) ? 4 : 2;     // log2(M/4)
    const long long tch = (long long)tot << cshift;
    edge_aggregate<<<(unsigned)((tch + 255) / 256), 256, 0, stream>>>(
        srcv, dstv, bufH, alpha, denom, bufA, Ee, Nn, Hh, M, cshift);
    // 6) BatchNorm + ReLU (conv bias cancels under BN, skipped)
    hipMemsetAsync(bnsum, 0, M * 4, stream);
    hipMemsetAsync(bnsq,  0, M * 4, stream);
    const int rpb = 128;
    bn_stats<<<(Nn + rpb - 1) / rpb, M, 0, stream>>>(bufA, bnsum, bnsq, Nn, M, rpb);
    const long long nm = (long long)Nn * M;
    bn_apply_relu<<<(unsigned)((nm + 255) / 256), 256, 0, stream>>>(
        bufA, bnsum, bnsq, bnG[li], bnB[li], outp, Nn, M);
  };

  run_layer(x,    128, 4, 64, 0, bufB);           // conv1 + bn1 + relu
  run_layer(bufB,  64, 4, 64, 1, bufB);           // conv2 + bn2 + relu (gemm consumes bufB first)
  run_layer(bufB,  64, 1, 16, 2, (float*)d_out);  // conv3 + bn3 + relu -> output
}
